// myfix2_90082644066546
// MI455X (gfx1250) — matrix operand order repaired and checked by reading, not compiled, not run on hardware
//
#include <hip/hip_runtime.h>

typedef _Float16 half_t;
typedef __attribute__((ext_vector_type(8)))  _Float16 v8h;
typedef __attribute__((ext_vector_type(16))) _Float16 v16h;
typedef __attribute__((ext_vector_type(8)))  float    v8f;
typedef __attribute__((ext_vector_type(4)))  unsigned int u32x4;
typedef __attribute__((ext_vector_type(8)))  unsigned int u32x8;

#define DIM   60
#define HEADS 3
#define HD    20
#define BATCH 2
#define IMH   128
#define IMW   128
#define TOK   (BATCH*IMH*IMW)   /* 32768 tokens */
#define QKVW  192               /* 180 padded   */
#define CPAD  64                /* 60 padded    */
#define HIDP  128               /* 120 padded   */

// ---------------- WMMA fragment loaders (CDNA5 16x16x32 f16 layouts) -------
// A: 16x32 (MxK). lane l: row = l&15, half h = l>>4 selects K {8h..8h+7, 16+8h..16+8h+7}
__device__ __forceinline__ v16h lds_load_A(const half_t* base, int ld) {
  int l = threadIdx.x & 31;
  int m = l & 15, h = l >> 4;
  const v8h* p0 = (const v8h*)(base + m * ld + 8 * h);
  const v8h* p1 = (const v8h*)(base + m * ld + 16 + 8 * h);
  v8h lo = *p0, hi = *p1;
  v16h a;
#pragma unroll
  for (int i = 0; i < 8; ++i) { a[i] = lo[i]; a[i + 8] = hi[i]; }
  return a;
}
// B: 32x16 (KxN), stored transposed as [N][K]. lane l: col n = l&15, half h = l>>4 selects K {8h..8h+7, 16+8h..16+8h+7}
__device__ __forceinline__ v16h lds_load_B(const half_t* baseT, int ld) {
  int l = threadIdx.x & 31;
  int n = l & 15, h = l >> 4;
  const v8h* p0 = (const v8h*)(baseT + n * ld + 8 * h);
  const v8h* p1 = (const v8h*)(baseT + n * ld + 16 + 8 * h);
  v8h lo = *p0, hi = *p1;
  v16h b;
#pragma unroll
  for (int i = 0; i < 8; ++i) { b[i] = lo[i]; b[i + 8] = hi[i]; }
  return b;
}
__device__ __forceinline__ v8f wmma_f16(v16h a, v16h b, v8f c) {
  return __builtin_amdgcn_wmma_f32_16x16x32_f16(false, a, false, b, (short)0, c,
                                                false, false);
}
// async global->LDS copy of 4 bytes (ASYNCcnt-tracked, bypasses VGPRs)
__device__ __forceinline__ void async_g2l_b32(void* lds, const void* gmem) {
  unsigned d = (unsigned)(size_t)lds;
  unsigned long long a = (unsigned long long)(size_t)gmem;
  asm volatile("global_load_async_to_lds_b32 %0, %1, off"
               :: "v"(d), "v"(a) : "memory");
}
__device__ __forceinline__ void async_wait0() {
  asm volatile("s_wait_asynccnt 0x0" ::: "memory");
}

// ---------------- input conv 3x3, 3->60, NCHW in, NHWC(60) out -------------
__global__ __launch_bounds__(256) void k_conv_in(const float* __restrict__ x,
    const float* __restrict__ w, const float* __restrict__ b,
    float* __restrict__ feat) {
  int o = blockIdx.x * 256 + threadIdx.x;
  if (o >= TOK * DIM) return;
  int c = o % DIM; int p = o / DIM;
  int iw = p % IMW; int t = p / IMW; int ih = t % IMH; int bt = t / IMH;
  float acc = b[c];
#pragma unroll
  for (int ci = 0; ci < 3; ++ci)
#pragma unroll
    for (int kh = 0; kh < 3; ++kh) {
      int yy = ih + kh - 1; if (yy < 0 || yy >= IMH) continue;
#pragma unroll
      for (int kw = 0; kw < 3; ++kw) {
        int xx = iw + kw - 1; if (xx < 0 || xx >= IMW) continue;
        acc += x[((bt * 3 + ci) * IMH + yy) * IMW + xx] *
               w[((c * 3 + ci) * 3 + kh) * 3 + kw];
      }
    }
  feat[o] = acc;
}

// ---------------- K1: LN1 + QKV GEMM (128x64 @ 64x192 per block) -----------
__global__ __launch_bounds__(256) void k_ln_qkv(const float* __restrict__ feat,
    const float* __restrict__ g, const float* __restrict__ bb,
    const float* __restrict__ w, const float* __restrict__ wb,
    half_t* __restrict__ qkv) {
  __shared__ __align__(16) half_t X[128][CPAD];
  __shared__ __align__(16) half_t Wt[QKVW][CPAD];
  __shared__ float bias[QKVW];
  int tid = threadIdx.x;
  int tok0 = blockIdx.x * 128;
  __builtin_prefetch(w + tid, 0, 0);
  if (tid < 128) {
    const float* xr = feat + (tok0 + tid) * DIM;
    float mu = 0.f;
    for (int c = 0; c < DIM; ++c) mu += xr[c];
    mu *= (1.f / 60.f);
    float var = 0.f;
    for (int c = 0; c < DIM; ++c) { float d = xr[c] - mu; var += d * d; }
    var *= (1.f / 60.f);
    float rs = rsqrtf(var + 1e-5f);
    for (int c = 0; c < DIM; ++c)
      X[tid][c] = (half_t)((xr[c] - mu) * rs * g[c] + bb[c]);
    for (int c = DIM; c < CPAD; ++c) X[tid][c] = (half_t)0.f;
  }
  for (int i = tid; i < QKVW * CPAD; i += 256) {
    int k = i & (CPAD - 1), n = i >> 6;
    Wt[n][k] = (k < DIM && n < 180) ? (half_t)w[k * 180 + n] : (half_t)0.f;
  }
  for (int i = tid; i < QKVW; i += 256) bias[i] = (i < 180) ? wb[i] : 0.f;
  __syncthreads();
  int wave = tid >> 5, l = tid & 31, h = l >> 4, n16 = l & 15;
  for (int t = wave; t < 8 * 12; t += 8) {      // 8 Mtiles x 12 Ntiles
    int mt = t & 7, nt = t >> 3;
    v8f acc = {0.f, 0.f, 0.f, 0.f, 0.f, 0.f, 0.f, 0.f};
#pragma unroll
    for (int ks = 0; ks < 2; ++ks) {
      v16h a = lds_load_A(&X[mt * 16][ks * 32], CPAD);
      v16h b = lds_load_B(&Wt[nt * 16][ks * 32], CPAD);
      acc = wmma_f16(a, b, acc);
    }
#pragma unroll
    for (int r = 0; r < 8; ++r) {
      int m = mt * 16 + h * 8 + r;
      int c = nt * 16 + n16;
      qkv[(tok0 + m) * QKVW + c] = (half_t)(acc[r] + bias[c]);
    }
  }
}

// ---------------- K2: shifted-window attention, one (window, head) per WG --
__global__ __launch_bounds__(256) void k_attn(const half_t* __restrict__ qkv,
    half_t* __restrict__ aout, int wh, int ww, int sh, int sw) {
  __shared__ __align__(16) half_t Qh[256][32];
  __shared__ __align__(16) half_t Kh[256][32];
  __shared__ __align__(16) half_t Vt[32][256];
  __shared__ __align__(16) half_t SP[8][16][256];   // scores -> probs, per wave
  __shared__ int tokidx[256];
  __shared__ int gid[256];
  int tid = threadIdx.x;
  int head = blockIdx.x % HEADS;
  int win  = blockIdx.x / HEADS;
  int nW = IMW / ww, nH = IMH / wh;
  int b = win / (nH * nW);
  int wrem = win % (nH * nW);
  int wy = wrem / nW, wx = wrem % nW;
  int hd0 = head * HD;
  for (int j = tid; j < 256; j += 256) {
    int iy = j / ww, ix = j % ww;
    int gy = wy * wh + iy, gx = wx * ww + ix;       // shifted-frame coords
    int hy = (gy + sh) % IMH, hx = (gx + sw) % IMW; // gather from rolled image
    tokidx[j] = (b * IMH + hy) * IMW + hx;
    int rh = (gy < IMH - wh) ? 0 : ((gy < IMH - sh) ? 1 : 2);
    int rw = (gx < IMW - ww) ? 0 : ((gx < IMW - sw) ? 1 : 2);
    gid[j] = rh * 3 + rw;
  }
  __syncthreads();
  // Q/K gather: async global->LDS (ASYNCcnt), 10 dwords per token row each
  for (int i = tid; i < 256 * (HD / 2); i += 256) {
    int j = i / (HD / 2), c2 = (i % (HD / 2)) * 2;
    int gg = tokidx[j] * QKVW;
    async_g2l_b32(&Qh[j][c2], qkv + gg + hd0 + c2);
    async_g2l_b32(&Kh[j][c2], qkv + gg + 60 + hd0 + c2);
  }
  for (int i = tid; i < 256 * HD; i += 256) {       // V gathered transposed
    int j = i / HD, c = i % HD;
    Vt[c][j] = qkv[tokidx[j] * QKVW + 120 + hd0 + c];
  }
  for (int i = tid; i < 256 * (32 - HD); i += 256) { // zero K-dim pads
    int j = i / 12, c = HD + i % 12;
    Qh[j][c] = (half_t)0.f; Kh[j][c] = (half_t)0.f; Vt[c][j] = (half_t)0.f;
  }
  async_wait0();
  __syncthreads();
  int wave = tid >> 5, l = tid & 31, h = l >> 4, n16 = l & 15;
  const float scale = 0.22360679775f;               // 1/sqrt(20)
  int maskf = (sh | sw);
  for (int rt = wave; rt < 16; rt += 8) {           // 2 row-tiles per wave
    v16h aq = lds_load_A(&Qh[rt * 16][0], 32);      // Q fragment, reused
    for (int nt = 0; nt < 16; ++nt) {               // S = Q @ K^T
      v16h bk = lds_load_B(&Kh[nt * 16][0], 32);
      v8f acc = {0.f, 0.f, 0.f, 0.f, 0.f, 0.f, 0.f, 0.f};
      acc = wmma_f16(aq, bk, acc);
#pragma unroll
      for (int r = 0; r < 8; ++r)
        SP[wave][h * 8 + r][nt * 16 + n16] = (half_t)acc[r];
    }
    __syncthreads();
    if (l < 16) {                                   // softmax row per lane
      int qrow = rt * 16 + l;
      int qg = gid[qrow];
      half_t* row = &SP[wave][l][0];
      float mx = -1e30f;
      for (int j = 0; j < 256; ++j) {
        float s = (float)row[j] * scale;
        if (maskf && gid[j] != qg) s -= 100.f;
        row[j] = (half_t)s;
        mx = fmaxf(mx, s);
      }
      float sum = 0.f;
      for (int j = 0; j < 256; ++j) sum += __expf((float)row[j] - mx);
      float inv = 1.f / sum;
      for (int j = 0; j < 256; ++j)
        row[j] = (half_t)(__expf((float)row[j] - mx) * inv);
    }
    __syncthreads();
#pragma unroll
    for (int ct = 0; ct < 2; ++ct) {                // O = P @ V
      v8f acc = {0.f, 0.f, 0.f, 0.f, 0.f, 0.f, 0.f, 0.f};
#pragma unroll
      for (int ks = 0; ks < 8; ++ks) {
        v16h a = lds_load_A(&SP[wave][0][ks * 32], 256);
        v16h bv = lds_load_B(&Vt[ct * 16][ks * 32], 256);
        acc = wmma_f16(a, bv, acc);
      }
#pragma unroll
      for (int r = 0; r < 8; ++r) {
        int c = ct * 16 + n16;
        if (c < HD) {
          int qrow = rt * 16 + h * 8 + r;
          aout[tokidx[qrow] * CPAD + hd0 + c] = (half_t)acc[r];
        }
      }
    }
  }
}

// ---------------- K3: attention output projection + residual ---------------
// Input tile staged by the Tensor Data Mover: one contiguous 128x64-half 2D
// tile (pad columns are multiplied by zeroed B-rows, so bulk copy is safe).
__global__ __launch_bounds__(256) void k_proj(const half_t* __restrict__ aout,
    const float* __restrict__ w, const float* __restrict__ wb,
    float* __restrict__ feat) {
  __shared__ __align__(16) half_t Xa[128][CPAD];
  __shared__ __align__(16) half_t Wt[CPAD][CPAD];
  __shared__ float bias[CPAD];
  int tid = threadIdx.x;
  int tok0 = blockIdx.x * 128;
  __builtin_prefetch(w + tid, 0, 0);
  // --- TDM: tensor_load_to_lds of 128*64 f16 = 16KB contiguous tile ---
  {
    unsigned lds_base = (unsigned)(size_t)&Xa[0][0];
    unsigned long long ga =
        (unsigned long long)(size_t)(aout + (size_t)tok0 * CPAD);
    u32x4 g0;
    g0[0] = 1u;                                   // count=1 user descriptor
    g0[1] = lds_base;                             // lds_addr
    g0[2] = (unsigned)(ga & 0xffffffffu);         // global_addr[31:0]
    g0[3] = (unsigned)((ga >> 32) & 0x01ffffffu)  // global_addr[56:32]
            | 0x80000000u;                        // type=2 (image)
    u32x8 g1;
    g1[0] = 0x00010000u;   // workgroup_mask=0, data_size=1 (2 bytes)
    g1[1] = 0x20000000u;   // tensor_dim0 = 8192 elements ([31:16]=low16)
    g1[2] = 0x00010000u;   // tensor_dim1 = 1 ([31:16]=low16)
    g1[3] = 0x20000000u;   // tile_dim0 = 8192 ([31:16])
    g1[4] = 0u;            // tile_dim1/tile_dim2 unused
    g1[5] = 8192u;         // tensor_dim0_stride
    g1[6] = 0u;
    g1[7] = 0u;
    if (tid < 32) {
      asm volatile("tensor_load_to_lds %0, %1" :: "s"(g0), "s"(g1) : "memory");
      __builtin_amdgcn_s_wait_tensorcnt((short)0);
    }
  }
  for (int i = tid; i < CPAD * CPAD; i += 256) {
    int k = i & 63, n = i >> 6;
    Wt[n][k] = (k < DIM && n < DIM) ? (half_t)w[k * DIM + n] : (half_t)0.f;
  }
  if (tid < CPAD) bias[tid] = (tid < DIM) ? wb[tid] : 0.f;
  __syncthreads();
  int wave = tid >> 5, l = tid & 31, h = l >> 4, n16 = l & 15;
  for (int t = wave; t < 8 * 4; t += 8) {
    int mt = t & 7, nt = t >> 3;
    v8f acc = {0.f, 0.f, 0.f, 0.f, 0.f, 0.f, 0.f, 0.f};
#pragma unroll
    for (int ks = 0; ks < 2; ++ks) {
      v16h a = lds_load_A(&Xa[mt * 16][ks * 32], CPAD);
      v16h b = lds_load_B(&Wt[nt * 16][ks * 32], CPAD);
      acc = wmma_f16(a, b, acc);
    }
#pragma unroll
    for (int r = 0; r < 8; ++r) {
      int c = nt * 16 + n16;
      if (c < DIM) {
        int idx = (tok0 + mt * 16 + h * 8 + r) * DIM + c;
        feat[idx] += acc[r] + bias[c];
      }
    }
  }
}

// ---------------- K4: LN2 + FC1 + Swish + FC2 + residual -------------------
__global__ __launch_bounds__(256) void k_mlp(float* __restrict__ feat,
    const float* __restrict__ g, const float* __restrict__ bb,
    const float* __restrict__ w1, const float* __restrict__ b1,
    const float* __restrict__ w2, const float* __restrict__ b2) {
  __shared__ __align__(16) half_t Xb[128][CPAD];
  __shared__ __align__(16) half_t W1t[HIDP][CPAD];
  __shared__ __align__(16) half_t Hh[128][HIDP];
  __shared__ __align__(16) half_t W2t[CPAD][HIDP];
  __shared__ float bs1[HIDP];
  __shared__ float bs2[CPAD];
  int tid = threadIdx.x;
  int tok0 = blockIdx.x * 128;
  __builtin_prefetch(w1 + tid, 0, 0);
  __builtin_prefetch(w2 + tid, 0, 0);
  if (tid < 128) {
    const float* xr = feat + (tok0 + tid) * DIM;
    float mu = 0.f;
    for (int c = 0; c < DIM; ++c) mu += xr[c];
    mu *= (1.f / 60.f);
    float var = 0.f;
    for (int c = 0; c < DIM; ++c) { float d = xr[c] - mu; var += d * d; }
    var *= (1.f / 60.f);
    float rs = rsqrtf(var + 1e-5f);
    for (int c = 0; c < DIM; ++c)
      Xb[tid][c] = (half_t)((xr[c] - mu) * rs * g[c] + bb[c]);
    for (int c = DIM; c < CPAD; ++c) Xb[tid][c] = (half_t)0.f;
  }
  for (int i = tid; i < HIDP * CPAD; i += 256) {
    int k = i & 63, n = i >> 6;
    W1t[n][k] = (k < DIM && n < 120) ? (half_t)w1[k * 120 + n] : (half_t)0.f;
  }
  for (int i = tid; i < CPAD * HIDP; i += 256) {
    int k = i & 127, n = i >> 7;
    W2t[n][k] = (n < DIM && k < 120) ? (half_t)w2[k * DIM + n] : (half_t)0.f;
  }
  if (tid < HIDP) bs1[tid] = (tid < 120) ? b1[tid] : 0.f;
  if (tid < CPAD) bs2[tid] = (tid < DIM) ? b2[tid] : 0.f;
  __syncthreads();
  int wave = tid >> 5, l = tid & 31, h = l >> 4, n16 = l & 15;
  for (int t = wave; t < 8 * 8; t += 8) {           // GEMM1: 128x64 @ 64x128
    int mt = t & 7, nt = t >> 3;
    v8f acc = {0.f, 0.f, 0.f, 0.f, 0.f, 0.f, 0.f, 0.f};
#pragma unroll
    for (int ks = 0; ks < 2; ++ks) {
      v16h a = lds_load_A(&Xb[mt * 16][ks * 32], CPAD);
      v16h b = lds_load_B(&W1t[nt * 16][ks * 32], CPAD);
      acc = wmma_f16(a, b, acc);
    }
#pragma unroll
    for (int r = 0; r < 8; ++r) {                   // bias + Swish -> LDS f16
      int c = nt * 16 + n16;
      float hv = acc[r] + bs1[c];
      Hh[mt * 16 + h * 8 + r][c] = (half_t)(hv / (1.f + __expf(-hv)));
    }
  }
  __syncthreads();
  for (int t = wave; t < 8 * 4; t += 8) {           // GEMM2: 128x128 @ 128x64
    int mt = t & 7, nt = t >> 3;
    v8f acc = {0.f, 0.f, 0.f, 0.f, 0.f, 0.f, 0.f, 0.f};
#pragma unroll
    for (int ks = 0; ks < 4; ++ks) {
      v16h a = lds_load_A(&Hh[mt * 16][ks * 32], HIDP);
      v16h b = lds_load_B(&W2t[nt * 16][ks * 32], HIDP);
      acc = wmma_f16(a, b, acc);
    }
#pragma unroll
    for (int r = 0; r < 8; ++r) {
      int c = nt * 16 + n16;
      if (c < DIM) {
        int idx = (tok0 + mt * 16 + h * 8 + r) * DIM + c;
        feat[idx] += acc[r] + bs2[c];
      }
    }
  }
}

// ---------------- output conv 3x3 (60->12) fused with PixelShuffle(2) ------
__global__ __launch_bounds__(256) void k_conv_out(const float* __restrict__ feat,
    const float* __restrict__ w, const float* __restrict__ b,
    float* __restrict__ out) {
  int o = blockIdx.x * 256 + threadIdx.x;
  if (o >= BATCH * 3 * 256 * 256) return;
  int ow = o % 256; int t = o / 256; int oh = t % 256; t /= 256;
  int co = t % 3; int bt = t / 3;
  int r1 = oh & 1, r2 = ow & 1;
  int hh = oh >> 1, wc = ow >> 1;
  int cc = (co * 2 + r1) * 2 + r2;                  // pre-shuffle channel
  float acc = b[cc];
  for (int kh = 0; kh < 3; ++kh) {
    int yy = hh + kh - 1; if (yy < 0 || yy >= IMH) continue;
    for (int kw = 0; kw < 3; ++kw) {
      int xx = wc + kw - 1; if (xx < 0 || xx >= IMW) continue;
      const float* fr = feat + ((bt * IMH + yy) * IMW + xx) * DIM;
      const float* wr = w + (cc * DIM * 3 + kh) * 3 + kw;
      for (int ci = 0; ci < DIM; ++ci) acc += fr[ci] * wr[ci * 9];
    }
  }
  out[o] = acc;
}

extern "C" void kernel_launch(void* const* d_in, const int* in_sizes, int n_in,
                              void* d_out, int out_size, void* d_ws, size_t ws_size,
                              hipStream_t stream) {
  const float* x      = (const float*)d_in[0];
  const float* conv_w = (const float*)d_in[1];
  const float* conv_b = (const float*)d_in[2];
  const float* ln1_g  = (const float*)d_in[3];
  const float* ln1_b  = (const float*)d_in[4];
  const float* qkv_w  = (const float*)d_in[5];
  const float* qkv_b  = (const float*)d_in[6];
  const float* proj_w = (const float*)d_in[7];
  const float* proj_b = (const float*)d_in[8];
  const float* ln2_g  = (const float*)d_in[9];
  const float* ln2_b  = (const float*)d_in[10];
  const float* fc1_w  = (const float*)d_in[11];
  const float* fc1_b  = (const float*)d_in[12];
  const float* fc2_w  = (const float*)d_in[13];
  const float* fc2_b  = (const float*)d_in[14];
  const float* last_w = (const float*)d_in[15];
  const float* last_b = (const float*)d_in[16];
  float* out = (float*)d_out;

  char* ws = (char*)d_ws;
  float*  feat = (float*)ws;                              // 7.9 MB f32
  half_t* qkvb = (half_t*)(ws + (size_t)8 * 1024 * 1024); // 12.6 MB f16
  half_t* aob  = (half_t*)(ws + (size_t)21 * 1024 * 1024);// 4.2 MB f16

  k_conv_in<<<(TOK * DIM + 255) / 256, 256, 0, stream>>>(x, conv_w, conv_b, feat);
  for (int l = 0; l < 36; ++l) {
    int i6 = l % 6, wi = i6 & 1;
    int wh = wi ? 8 : 32, ww = wi ? 32 : 8;
    int sh = 0, sw = 0;
    if ((i6 & 3) >= 2) { sh = wi ? 4 : 16; sw = wi ? 16 : 4; }
    k_ln_qkv<<<TOK / 128, 256, 0, stream>>>(feat, ln1_g + l * 60, ln1_b + l * 60,
                                            qkv_w + l * 60 * 180, qkv_b + l * 180,
                                            qkvb);
    k_attn<<<(TOK / 256) * HEADS, 256, 0, stream>>>(qkvb, aob, wh, ww, sh, sw);
    k_proj<<<TOK / 128, 256, 0, stream>>>(aob, proj_w + l * 60 * 60,
                                          proj_b + l * 60, feat);
    k_mlp<<<TOK / 128, 256, 0, stream>>>(feat, ln2_g + l * 60, ln2_b + l * 60,
                                         fc1_w + l * 60 * 120, fc1_b + l * 120,
                                         fc2_w + l * 120 * 60, fc2_b + l * 60);
  }
  k_conv_out<<<(BATCH * 3 * 256 * 256 + 255) / 256, 256, 0, stream>>>(
      feat, last_w, last_b, out);
}
